// SplitLSTM_66881230734048
// MI455X (gfx1250) — compile-verified, matrix operand order checked by reading, not run on hardware
//
#include <hip/hip_runtime.h>
#include <hip/hip_bf16.h>

// SplitLSTM for MI455X (gfx1250): bf16 WMMA recurrent kernel.
// All operands (x, h, weights) are pre-converted/packed to bf16 once, so the
// per-timestep GEMM inner loops are pure global_load_b128 + v_wmma.

#define BSZ   64
#define TLEN  512
#define NSPL  4
#define INPD  256
#define HIDD  256
#define HDIM  1024

typedef __attribute__((ext_vector_type(4)))  float  v4f;
typedef __attribute__((ext_vector_type(8)))  float  v8f;
typedef __attribute__((ext_vector_type(16))) __bf16 v16bf;

__device__ __forceinline__ unsigned short f2bf_u16(float f) {
  union { float f; unsigned u; } v; v.f = f;
  unsigned r = (v.u + 0x7FFFu + ((v.u >> 16) & 1u)) >> 16;   // round-to-nearest-even
  return (unsigned short)r;
}

// A-fragment (16x32 bf16) from row-major bf16 memory: two 16B chunks per lane.
__device__ __forceinline__ v16bf load_a_bf(const unsigned short* p0,
                                           const unsigned short* p1) {
  union { v8f v; v4f q[2]; } u;
  u.q[0] = *(const v4f*)p0;
  u.q[1] = *(const v4f*)p1;
  return __builtin_bit_cast(v16bf, u.v);
}

// B-fragment from packed weights: 32 contiguous bytes per lane.
__device__ __forceinline__ v16bf load_b_pk(const unsigned short* p) {
  union { v8f v; v4f q[2]; } u;
  u.q[0] = *(const v4f*)p;
  u.q[1] = *(const v4f*)(p + 8);
  return __builtin_bit_cast(v16bf, u.v);
}

__device__ __forceinline__ v8f wmma_bf16(v16bf a, v16bf b, v8f c) {
  return __builtin_amdgcn_wmma_f32_16x16x32_bf16(false, a, false, b,
                                                 (short)0, c, false, false);
}

__device__ __forceinline__ float sigmoid_f(float x) {
  return 1.0f / (1.0f + __expf(-x));
}
__device__ __forceinline__ float tanh_f(float x) {
  x = fminf(fmaxf(x, -15.0f), 15.0f);
  float e = __expf(-2.0f * x);
  return (1.0f - e) / (1.0f + e);
}

// ---------------------------------------------------------------------------
// One-shot f32 -> bf16 conversion of x (4 elements / thread, b128 in, b64 out)
// ---------------------------------------------------------------------------
__global__ __launch_bounds__(256) void convert_x_kernel(
    const float* __restrict__ x, unsigned short* __restrict__ xbf, long n4) {
  long i = (long)blockIdx.x * blockDim.x + threadIdx.x;
  if (i >= n4) return;
  v4f v = ((const v4f*)x)[i];
  union { unsigned short s[4]; unsigned long long u; } o;
  o.s[0] = f2bf_u16(v[0]);
  o.s[1] = f2bf_u16(v[1]);
  o.s[2] = f2bf_u16(v[2]);
  o.s[3] = f2bf_u16(v[3]);
  ((unsigned long long*)xbf)[i] = o.u;
}

// ---------------------------------------------------------------------------
// Pack a weight stack W[S][K][N] (f32, row-major) into bf16 WMMA B-fragments:
// dst index = (((s*NT + nt)*KB + kb)*32 + lane)*16 + j, holding W[k][n] with
// n = nt*16 + (lane&15), k = kb*32 + ((lane>>4)*8 + j)  (j<8)
//                      = kb*32 + 16 + (lane>>4)*8 + j-8 (j>=8)
// ---------------------------------------------------------------------------
__global__ __launch_bounds__(256) void pack_weight_kernel(
    const float* __restrict__ W, unsigned short* __restrict__ Wpk,
    int K, int N, long total) {
  long i = (long)blockIdx.x * blockDim.x + threadIdx.x;
  if (i >= total) return;
  int  j    = (int)(i & 15);
  long r    = i >> 4;
  int  lane = (int)(r & 31); r >>= 5;
  int  KB   = K >> 5;
  int  kb   = (int)(r % KB); r /= KB;
  int  NT   = N >> 4;
  int  nt   = (int)(r % NT);
  long s    = r / NT;
  int  n    = nt * 16 + (lane & 15);
  int  cs   = lane >> 4;
  int  k    = kb * 32 + ((j < 8) ? (cs * 8 + j) : (16 + cs * 8 + (j - 8)));
  Wpk[i] = f2bf_u16(W[(s * K + k) * (long)N + n]);
}

__global__ __launch_bounds__(256) void init_state_kernel(
    unsigned short* __restrict__ h_bf, float* __restrict__ c) {
  int i = blockIdx.x * 256 + threadIdx.x;
  if (i < BSZ * HDIM) { h_bf[i] = 0; c[i] = 0.0f; }
}

// ---------------------------------------------------------------------------
// Step A: per-split gate pre-activations.
//   f: xs[64,256] @ Wf[s] (K=256);  i/g/o: [xs|hs][64,512] @ W[s] (K=512)
// Grid 128 x 128 (512 waves). Wave -> (gate, split, mtile, 2 n-tiles).
// Output: preact[4][64][1024] bf16 (bias added here).
// ---------------------------------------------------------------------------
__global__ __launch_bounds__(128) void gates_kernel(
    const unsigned short* __restrict__ xbf,    // [B,T,1024] bf16
    const unsigned short* __restrict__ h_bf,   // [64,1024]  bf16
    const unsigned short* __restrict__ WfT,    // [4][16nt][ 8kb][32][16]
    const unsigned short* __restrict__ WiT,    // [4][16nt][16kb][32][16]
    const unsigned short* __restrict__ WgT,
    const unsigned short* __restrict__ WoT,
    const float* __restrict__ bfv, const float* __restrict__ biv,
    const float* __restrict__ bgv, const float* __restrict__ bov,
    unsigned short* __restrict__ preact,       // [4][64][1024]
    int t) {
  int lane = threadIdx.x & 31;
  int w    = blockIdx.x * 4 + (threadIdx.x >> 5);   // 0..511
  int gate = w >> 7;
  int rem  = w & 127;
  int s     = rem >> 5;
  int r2    = rem & 31;
  int mtile = r2 >> 3;
  int nt0   = (r2 & 7) * 2;                         // 2 n-tiles / wave
  int m  = lane & 15;
  int cs = lane >> 4;
  int brow_base = mtile * 16;

  const unsigned short* Wbase;
  int KB;
  if (gate == 0)      { Wbase = WfT + (long)s * (16 * 8  * 32 * 16); KB = 8;  }
  else if (gate == 1) { Wbase = WiT + (long)s * (16 * 16 * 32 * 16); KB = 16; }
  else if (gate == 2) { Wbase = WgT + (long)s * (16 * 16 * 32 * 16); KB = 16; }
  else                { Wbase = WoT + (long)s * (16 * 16 * 32 * 16); KB = 16; }

  const unsigned short* xrow =
      xbf + (((long)(brow_base + m) * TLEN + t) << 10) + s * INPD;

  v8f acc0 = {}, acc1 = {};

  // x part: K = 0..255 (kb 0..7)
#pragma unroll
  for (int kb = 0; kb < 8; ++kb) {
    int k0 = kb * 32 + cs * 8;
    v16bf a = load_a_bf(xrow + k0, xrow + k0 + 16);
    const unsigned short* bp0 = Wbase + ((((long)(nt0    ) * KB + kb) * 32 + lane) << 4);
    const unsigned short* bp1 = Wbase + ((((long)(nt0 + 1) * KB + kb) * 32 + lane) << 4);
    acc0 = wmma_bf16(a, load_b_pk(bp0), acc0);
    acc1 = wmma_bf16(a, load_b_pk(bp1), acc1);
  }
  // h part for i/g/o: K = 256..511 (kb 8..15)
  if (gate != 0) {
    const unsigned short* hrow = h_bf + (brow_base + m) * HDIM + s * HIDD;
#pragma unroll
    for (int kb = 8; kb < 16; ++kb) {
      int k0 = (kb - 8) * 32 + cs * 8;
      v16bf a = load_a_bf(hrow + k0, hrow + k0 + 16);
      const unsigned short* bp0 = Wbase + ((((long)(nt0    ) * KB + kb) * 32 + lane) << 4);
      const unsigned short* bp1 = Wbase + ((((long)(nt0 + 1) * KB + kb) * 32 + lane) << 4);
      acc0 = wmma_bf16(a, load_b_pk(bp0), acc0);
      acc1 = wmma_bf16(a, load_b_pk(bp1), acc1);
    }
  }

  const float* bias = (gate == 0) ? bfv : (gate == 1) ? biv
                    : (gate == 2) ? bgv : bov;          // [NS,256]
  v8f acc[2] = {acc0, acc1};
#pragma unroll
  for (int q = 0; q < 2; ++q) {
    int colg = s * HIDD + (nt0 + q) * 16 + m;           // global column in H
    float bv = bias[colg];
#pragma unroll
    for (int vg = 0; vg < 8; ++vg) {
      int brow = brow_base + cs * 8 + vg;
      preact[(((long)gate * BSZ + brow) << 10) + colg] =
          f2bf_u16(acc[q][vg] + bv);
    }
  }
}

// ---------------------------------------------------------------------------
// Step B1: connectors. G[k] = preact[k][64,1024] @ Wc[k][1024,1024].
// Grid 128 x 128 (512 waves). Wave -> (gate, mtile, 2 n-tiles), K-loop of 32.
// ---------------------------------------------------------------------------
__global__ __launch_bounds__(128) void connector_kernel(
    const unsigned short* __restrict__ preact,  // [4][64][1024] bf16
    const unsigned short* __restrict__ WcT,     // [4][64nt][32kb][32][16]
    float* __restrict__ G) {                    // [4][64][1024] f32
  int lane = threadIdx.x & 31;
  int w    = blockIdx.x * 4 + (threadIdx.x >> 5);  // 0..511
  int gate  = w >> 7;
  int rem   = w & 127;
  int mtile = rem >> 5;
  int nt0   = (rem & 31) * 2;
  int m  = lane & 15;
  int cs = lane >> 4;

  const unsigned short* arow =
      preact + (((long)gate * BSZ + mtile * 16 + m) << 10);
  const unsigned short* Wb = WcT + ((long)gate << 20);   // 1024*1024 per gate

  v8f acc0 = {}, acc1 = {};
#pragma unroll 4
  for (int kb = 0; kb < 32; ++kb) {
    int k0 = kb * 32 + cs * 8;
    v16bf a = load_a_bf(arow + k0, arow + k0 + 16);
    const unsigned short* bp0 = Wb + ((((long)(nt0    ) * 32 + kb) * 32 + lane) << 4);
    const unsigned short* bp1 = Wb + ((((long)(nt0 + 1) * 32 + kb) * 32 + lane) << 4);
    acc0 = wmma_bf16(a, load_b_pk(bp0), acc0);
    acc1 = wmma_bf16(a, load_b_pk(bp1), acc1);
  }

  v8f acc[2] = {acc0, acc1};
#pragma unroll
  for (int q = 0; q < 2; ++q) {
    int col = (nt0 + q) * 16 + m;
#pragma unroll
    for (int vg = 0; vg < 8; ++vg) {
      int row = mtile * 16 + cs * 8 + vg;
      G[(((long)gate * BSZ + row) << 10) + col] = acc[q][vg];
    }
  }
}

// ---------------------------------------------------------------------------
// Step B2: LSTM nonlinearities + state update; writes h to d_out[:,t,:].
// ---------------------------------------------------------------------------
__global__ __launch_bounds__(256) void update_kernel(
    const float* __restrict__ G,     // [4][64][1024]
    const float* __restrict__ bc,    // [4][1024]
    float* __restrict__ c,           // [64,1024]
    unsigned short* __restrict__ h_bf,
    float* __restrict__ out,         // [B,T,H]
    int t) {
  int i = blockIdx.x * 256 + threadIdx.x;
  if (i >= BSZ * HDIM) return;
  int b = i >> 10, col = i & 1023;
  const int STRIDE = BSZ * HDIM;
  float f  = sigmoid_f(G[i]              + bc[col]);
  float ii = sigmoid_f(G[STRIDE + i]     + bc[HDIM + col]);
  float g  = tanh_f   (G[2 * STRIDE + i] + bc[2 * HDIM + col]);
  float o  = sigmoid_f(G[3 * STRIDE + i] + bc[3 * HDIM + col]);
  float cn = f * c[i] + ii * g;
  float hn = o * tanh_f(cn);
  c[i]    = cn;
  h_bf[i] = f2bf_u16(hn);
  out[((long)b * TLEN + t) * HDIM + col] = hn;
}

// ---------------------------------------------------------------------------
extern "C" void kernel_launch(void* const* d_in, const int* in_sizes, int n_in,
                              void* d_out, int out_size, void* d_ws, size_t ws_size,
                              hipStream_t stream) {
  const float* x   = (const float*)d_in[0];
  const float* Wf  = (const float*)d_in[1];
  const float* bfv = (const float*)d_in[2];
  const float* Wi  = (const float*)d_in[3];
  const float* biv = (const float*)d_in[4];
  const float* Wg  = (const float*)d_in[5];
  const float* bgv = (const float*)d_in[6];
  const float* Wo  = (const float*)d_in[7];
  const float* bov = (const float*)d_in[8];
  const float* Wc  = (const float*)d_in[9];
  const float* bc  = (const float*)d_in[10];
  float* out = (float*)d_out;

  // Workspace layout (all region sizes are 16B multiples)
  unsigned short* WfT    = (unsigned short*)d_ws;          //  4*256*256
  unsigned short* WiT    = WfT + 262144;                   //  4*512*256
  unsigned short* WgT    = WiT + 524288;
  unsigned short* WoT    = WgT + 524288;
  unsigned short* WcT    = WoT + 524288;                   //  4*1024*1024
  unsigned short* preact = WcT + 4194304;                  //  4*64*1024
  unsigned short* h_bf   = preact + 262144;                //  64*1024
  float*          Gbuf   = (float*)(h_bf + 65536);         //  4*64*1024
  float*          cbuf   = Gbuf + 262144;                  //  64*1024
  unsigned short* xbf    = (unsigned short*)(cbuf + 65536);//  64*512*1024 bf16
  (void)in_sizes; (void)n_in; (void)out_size; (void)ws_size;

  long tot;
  tot = 4L * 256 * 256;
  pack_weight_kernel<<<(unsigned)((tot + 255) / 256), 256, 0, stream>>>(Wf, WfT, 256, 256, tot);
  tot = 4L * 512 * 256;
  pack_weight_kernel<<<(unsigned)((tot + 255) / 256), 256, 0, stream>>>(Wi, WiT, 512, 256, tot);
  pack_weight_kernel<<<(unsigned)((tot + 255) / 256), 256, 0, stream>>>(Wg, WgT, 512, 256, tot);
  pack_weight_kernel<<<(unsigned)((tot + 255) / 256), 256, 0, stream>>>(Wo, WoT, 512, 256, tot);
  tot = 4L * 1024 * 1024;
  pack_weight_kernel<<<(unsigned)((tot + 255) / 256), 256, 0, stream>>>(Wc, WcT, 1024, 1024, tot);

  long n4 = (long)BSZ * TLEN * HDIM / 4;                   // 8388608
  convert_x_kernel<<<(unsigned)((n4 + 255) / 256), 256, 0, stream>>>(x, xbf, n4);

  init_state_kernel<<<(BSZ * HDIM + 255) / 256, 256, 0, stream>>>(h_bf, cbuf);

  for (int t = 0; t < TLEN; ++t) {
    gates_kernel<<<128, 128, 0, stream>>>(xbf, h_bf, WfT, WiT, WgT, WoT,
                                          bfv, biv, bgv, bov, preact, t);
    connector_kernel<<<128, 128, 0, stream>>>(preact, WcT, Gbuf);
    update_kernel<<<(BSZ * HDIM + 255) / 256, 256, 0, stream>>>(Gbuf, bc, cbuf,
                                                                h_bf, out, t);
  }
}
